// S6_2680059593332
// MI455X (gfx1250) — matrix operand order repaired and checked by reading, not compiled, not run on hardware
//
#include <hip/hip_runtime.h>
#include <hip/hip_bf16.h>

// ---------------------------------------------------------------------------
// S6 (Mamba selective-scan block) for gfx1250 / MI455X.
//   b=2, l=1024, d=1024, dt_rank=64, n=16; M = b*l = 2048
//
// Strategy:
//   * One-shot prep: x -> bf16 row-major; W_x/W_dt/W_out -> bf16 transposed
//     (N-major) via LDS-tiled transpose-convert. After that every WMMA
//     fragment (A and B) is two contiguous global_load_b128's, no VALU.
//   * GEMMs on v_wmma_f32_16x16x32_bf16, 2x2 (or 2x1) register tiles per
//     wave so fragments are reused and WMMA issue density is high.
//   * Scan: fp32, one lane per (channel,state), v_exp_f32 recurrence,
//     16-lane shfl_xor reduction; writes y directly as bf16 for the big GEMM.
// ---------------------------------------------------------------------------

typedef __attribute__((ext_vector_type(16))) __bf16 v16bf;
typedef __attribute__((ext_vector_type(8)))  float  v8f;

#define M_ROWS 2048
#define D_DIM  1024
#define N_ST   16
#define DTRANK 64
#define E_COLS 96   // dt_rank + 2n

union FragBF {
    v16bf v;
    unsigned short u[16];
};

__device__ __forceinline__ unsigned short f2bf(float f) {
    unsigned int b = __builtin_bit_cast(unsigned int, f);
    b += 0x7fffu + ((b >> 16) & 1u);   // round-to-nearest-even
    return (unsigned short)(b >> 16);
}

// A-fragment (16x32, MxK) from bf16 row-major src.
// Lane (l16, half): row M=l16. e=0..7 -> K=kbase+8*half+e ; e=8..15 -> +16.
// Two contiguous 16B chunks -> 2x global_load_b128.
__device__ __forceinline__ v16bf load_a_bf16(const unsigned short* __restrict__ src,
                                             int row, int lda, int kbase, int half) {
    FragBF fa;
    const unsigned short* p = src + (size_t)row * lda + kbase + 8 * half;
    __builtin_memcpy(&fa.u[0], p, 16);
    __builtin_memcpy(&fa.u[8], p + 16, 16);
    return fa.v;
}

// B-fragment (32x16, KxN) from bf16 *transposed* (N-major) src: srcT[n*ldk + k].
// Lane holds column N=l16; e=0..7 -> K=kbase+8*half+e ; e=8..15 -> +16.
__device__ __forceinline__ v16bf load_bT_bf16(const unsigned short* __restrict__ srcT,
                                              int col, int ldk, int kbase, int half) {
    FragBF fb;
    const unsigned short* p = srcT + (size_t)col * ldk + kbase + 8 * half;
    __builtin_memcpy(&fb.u[0], p, 16);
    __builtin_memcpy(&fb.u[8], p + 16, 16);
    return fb.v;
}

__device__ __forceinline__ float softplus_f(float v) {
    if (v > 20.f) return v;
    return __logf(1.f + __expf(v));
}

#define WMMA_BF16(A, B, C) \
    __builtin_amdgcn_wmma_f32_16x16x32_bf16(false, (A), false, (B), (short)0, (C), false, false)

// ---------------------------------------------------------------------------
// Prep 1: elementwise f32 -> bf16, 8 elements/thread (b128 in, b128 out).
// ---------------------------------------------------------------------------
__global__ __launch_bounds__(256) void cvt_f32_bf16(const float* __restrict__ in,
                                                    unsigned short* __restrict__ out) {
    const size_t i = ((size_t)blockIdx.x * 256 + threadIdx.x) * 8;
    float v[8];
    __builtin_memcpy(&v[0], in + i, 16);
    __builtin_memcpy(&v[4], in + i + 4, 16);
    unsigned short o[8];
#pragma unroll
    for (int e = 0; e < 8; ++e) o[e] = f2bf(v[e]);
    __builtin_memcpy(out + i, o, 16);
}

// ---------------------------------------------------------------------------
// Prep 2: transpose + convert: in[k*N + n] (f32, KxN) -> out[n*K + k] (bf16).
// 32x32 LDS tile, both global sides coalesced. K, N multiples of 32.
// ---------------------------------------------------------------------------
__global__ __launch_bounds__(256) void transpose_cvt(const float* __restrict__ in,
                                                     unsigned short* __restrict__ out,
                                                     int K, int N) {
    __shared__ float tile[32][33];
    const int tx = threadIdx.x & 31;
    const int ty = threadIdx.x >> 5;          // 0..7
    const int n0 = blockIdx.x * 32;
    const int k0 = blockIdx.y * 32;
#pragma unroll
    for (int j = 0; j < 4; ++j)
        tile[ty + j * 8][tx] = in[(size_t)(k0 + ty + j * 8) * N + n0 + tx];
    __syncthreads();
#pragma unroll
    for (int j = 0; j < 4; ++j)
        out[(size_t)(n0 + ty + j * 8) * K + k0 + tx] = f2bf(tile[tx][ty + j * 8]);
}

// ---------------------------------------------------------------------------
// Kernel 1: x_dbl[2048,96] = x @ W_x.  2 M-tiles x 1 N-tile per wave.
// Also emits bf16 copy of cols 0..63 (the delta_raw block) for the next GEMM.
// ---------------------------------------------------------------------------
__global__ __launch_bounds__(128) void s6_proj(const unsigned short* __restrict__ xbf,
                                               const unsigned short* __restrict__ WxT,
                                               float* __restrict__ xdbl,
                                               unsigned short* __restrict__ xdblbf) {
    const int lane = threadIdx.x & 31;
    const int wave = threadIdx.x >> 5;
    const int half = lane >> 4, l16 = lane & 15;
    const int mp = blockIdx.x * 4 + wave;   // 0..63  (pair of M tiles)
    const int nt = blockIdx.y;              // 0..5
    const int row0 = mp * 32 + l16;
    const int col  = nt * 16 + l16;

    v8f acc0 = {}, acc1 = {};
    for (int kb = 0; kb < D_DIM; kb += 32) {
        v16bf a0 = load_a_bf16(xbf, row0,      D_DIM, kb, half);
        v16bf a1 = load_a_bf16(xbf, row0 + 16, D_DIM, kb, half);
        v16bf b  = load_bT_bf16(WxT, col, D_DIM, kb, half);
        acc0 = WMMA_BF16(a0, b, acc0);
        acc1 = WMMA_BF16(a1, b, acc1);
    }
#pragma unroll
    for (int r = 0; r < 8; ++r) {
        const int r0 = mp * 32 + r + 8 * half;
        const int r1 = r0 + 16;
        xdbl[(size_t)r0 * E_COLS + col] = acc0[r];
        xdbl[(size_t)r1 * E_COLS + col] = acc1[r];
        if (nt < 4) {   // delta_raw block -> bf16 for s6_delta A-fragments
            xdblbf[(size_t)r0 * DTRANK + col] = f2bf(acc0[r]);
            xdblbf[(size_t)r1 * DTRANK + col] = f2bf(acc1[r]);
        }
    }
}

// ---------------------------------------------------------------------------
// Kernel 2: delta = softplus(delta_raw @ W_dt + b_dt).  2x2 tiles per wave.
// ---------------------------------------------------------------------------
__global__ __launch_bounds__(256) void s6_delta(const unsigned short* __restrict__ xdblbf,
                                                const unsigned short* __restrict__ WdtT,
                                                const float* __restrict__ bdt,
                                                float* __restrict__ delta) {
    const int lane = threadIdx.x & 31;
    const int wave = threadIdx.x >> 5;
    const int half = lane >> 4, l16 = lane & 15;
    const int mp = blockIdx.x * 8 + wave;   // 0..63
    const int np = blockIdx.y;              // 0..31
    const int row0 = mp * 32 + l16;
    const int col0 = np * 32 + l16;

    v8f acc00 = {}, acc01 = {}, acc10 = {}, acc11 = {};
#pragma unroll
    for (int kb = 0; kb < DTRANK; kb += 32) {
        v16bf a0 = load_a_bf16(xdblbf, row0,      DTRANK, kb, half);
        v16bf a1 = load_a_bf16(xdblbf, row0 + 16, DTRANK, kb, half);
        v16bf b0 = load_bT_bf16(WdtT, col0,      DTRANK, kb, half);
        v16bf b1 = load_bT_bf16(WdtT, col0 + 16, DTRANK, kb, half);
        acc00 = WMMA_BF16(a0, b0, acc00);
        acc01 = WMMA_BF16(a0, b1, acc01);
        acc10 = WMMA_BF16(a1, b0, acc10);
        acc11 = WMMA_BF16(a1, b1, acc11);
    }
    const float bias0 = bdt[col0];
    const float bias1 = bdt[col0 + 16];
#pragma unroll
    for (int r = 0; r < 8; ++r) {
        const int r0 = mp * 32 + r + 8 * half;
        const int r1 = r0 + 16;
        delta[(size_t)r0 * D_DIM + col0]      = softplus_f(acc00[r] + bias0);
        delta[(size_t)r0 * D_DIM + col0 + 16] = softplus_f(acc01[r] + bias1);
        delta[(size_t)r1 * D_DIM + col0]      = softplus_f(acc10[r] + bias0);
        delta[(size_t)r1 * D_DIM + col0 + 16] = softplus_f(acc11[r] + bias1);
    }
}

// ---------------------------------------------------------------------------
// Kernel 3: selective scan. One lane per (channel, state): 2048*16 threads.
//   h <- exp(delta*A[n])*h + delta*u*B[t,n] ; y = sum_n h*C[t,n] + u*D
// ---------------------------------------------------------------------------
__global__ __launch_bounds__(256) void s6_scan(const float* __restrict__ x,
                                               const float* __restrict__ xdbl,
                                               const float* __restrict__ Alog,
                                               const float* __restrict__ Dp,
                                               const float* __restrict__ delta,
                                               unsigned short* __restrict__ ybf) {
    const int n = threadIdx.x & 15;                        // state index
    const int c = blockIdx.x * 16 + (threadIdx.x >> 4);    // channel 0..2047
    const int b = c >> 10;
    const int d = c & (D_DIM - 1);

    const float An = -__expf(Alog[d * N_ST + n]);
    const float Dd = Dp[d];
    float h = 0.f;

    const int rbase = b * 1024;
    for (int t = 0; t < 1024; ++t) {
        const int r = rbase + t;
        const float dl = delta[(size_t)r * D_DIM + d];
        const float u  = x[(size_t)r * D_DIM + d];
        const float Bv = xdbl[(size_t)r * E_COLS + DTRANK + n];
        const float Cv = xdbl[(size_t)r * E_COLS + DTRANK + N_ST + n];

        h = __expf(dl * An) * h + (dl * u) * Bv;
        float y = h * Cv;
#pragma unroll
        for (int off = 8; off >= 1; off >>= 1)
            y += __shfl_xor(y, off, 16);
        if (n == 0)
            ybf[(size_t)r * D_DIM + d] = f2bf(y + u * Dd);
    }
}

// ---------------------------------------------------------------------------
// Kernel 4: out = y @ W_out + b_out.  2x2 tiles per wave, all-b128 fragments.
// ---------------------------------------------------------------------------
__global__ __launch_bounds__(256) void s6_out(const unsigned short* __restrict__ ybf,
                                              const unsigned short* __restrict__ WoutT,
                                              const float* __restrict__ bout,
                                              float* __restrict__ out) {
    const int lane = threadIdx.x & 31;
    const int wave = threadIdx.x >> 5;
    const int half = lane >> 4, l16 = lane & 15;
    const int mp = blockIdx.x * 8 + wave;   // 0..63
    const int np = blockIdx.y;              // 0..31
    const int row0 = mp * 32 + l16;
    const int col0 = np * 32 + l16;

    v8f acc00 = {}, acc01 = {}, acc10 = {}, acc11 = {};
    for (int kb = 0; kb < D_DIM; kb += 32) {
        v16bf a0 = load_a_bf16(ybf, row0,      D_DIM, kb, half);
        v16bf a1 = load_a_bf16(ybf, row0 + 16, D_DIM, kb, half);
        v16bf b0 = load_bT_bf16(WoutT, col0,      D_DIM, kb, half);
        v16bf b1 = load_bT_bf16(WoutT, col0 + 16, D_DIM, kb, half);
        acc00 = WMMA_BF16(a0, b0, acc00);
        acc01 = WMMA_BF16(a0, b1, acc01);
        acc10 = WMMA_BF16(a1, b0, acc10);
        acc11 = WMMA_BF16(a1, b1, acc11);
    }
    const float bias0 = bout[col0];
    const float bias1 = bout[col0 + 16];
#pragma unroll
    for (int r = 0; r < 8; ++r) {
        const int r0 = mp * 32 + r + 8 * half;
        const int r1 = r0 + 16;
        out[(size_t)r0 * D_DIM + col0]      = acc00[r] + bias0;
        out[(size_t)r0 * D_DIM + col0 + 16] = acc01[r] + bias1;
        out[(size_t)r1 * D_DIM + col0]      = acc10[r] + bias0;
        out[(size_t)r1 * D_DIM + col0 + 16] = acc11[r] + bias1;
    }
}

// ---------------------------------------------------------------------------
extern "C" void kernel_launch(void* const* d_in, const int* in_sizes, int n_in,
                              void* d_out, int out_size, void* d_ws, size_t ws_size,
                              hipStream_t stream) {
    const float* x    = (const float*)d_in[0];   // (2,1024,1024)
    const float* Wx   = (const float*)d_in[1];   // (1024,96)
    const float* Wdt  = (const float*)d_in[2];   // (64,1024)
    const float* bdt  = (const float*)d_in[3];   // (1024,)
    const float* Alog = (const float*)d_in[4];   // (1024,16)
    const float* Dp   = (const float*)d_in[5];   // (1024,)
    const float* Wout = (const float*)d_in[6];   // (1024,1024)
    const float* bout = (const float*)d_in[7];   // (1024,)
    float* out = (float*)d_out;

    // Workspace layout (all offsets 256B-aligned):
    const size_t OFF_XDBL   = 0;          //  f32 2048*96    =   786,432 B
    const size_t OFF_DELTA  = 786432;     //  f32 2048*1024  = 8,388,608 B
    const size_t OFF_Y      = 9175040;    // bf16 2048*1024  = 4,194,304 B
    const size_t OFF_XBF    = 13369344;   // bf16 2048*1024  = 4,194,304 B
    const size_t OFF_XDBLBF = 17563648;   // bf16 2048*64    =   262,144 B
    const size_t OFF_WXT    = 17825792;   // bf16 96*1024    =   196,608 B
    const size_t OFF_WDTT   = 18022400;   // bf16 1024*64    =   131,072 B
    const size_t OFF_WOUTT  = 18153472;   // bf16 1024*1024  = 2,097,152 B
    const size_t WS_NEED    = 20250624;
    if (ws_size < WS_NEED) return;

    char* ws = (char*)d_ws;
    float*          xdbl   = (float*)(ws + OFF_XDBL);
    float*          delta  = (float*)(ws + OFF_DELTA);
    unsigned short* ybf    = (unsigned short*)(ws + OFF_Y);
    unsigned short* xbf    = (unsigned short*)(ws + OFF_XBF);
    unsigned short* xdblbf = (unsigned short*)(ws + OFF_XDBLBF);
    unsigned short* WxT    = (unsigned short*)(ws + OFF_WXT);
    unsigned short* WdtT   = (unsigned short*)(ws + OFF_WDTT);
    unsigned short* WoutT  = (unsigned short*)(ws + OFF_WOUTT);

    // --- one-shot conversions / transposes (L2-resident, ~11 MB total) ---
    cvt_f32_bf16 <<<dim3(1024),   256, 0, stream>>>(x, xbf);          // 2M elems
    transpose_cvt<<<dim3(3, 32),  256, 0, stream>>>(Wx,   WxT,   D_DIM,  E_COLS);
    transpose_cvt<<<dim3(32, 2),  256, 0, stream>>>(Wdt,  WdtT,  DTRANK, D_DIM);
    transpose_cvt<<<dim3(32, 32), 256, 0, stream>>>(Wout, WoutT, D_DIM,  D_DIM);

    // --- pipeline ---
    s6_proj <<<dim3(16, 6),  128, 0, stream>>>(xbf, WxT, xdbl, xdblbf);
    s6_delta<<<dim3(8, 32),  256, 0, stream>>>(xdblbf, WdtT, bdt, delta);
    s6_scan <<<dim3(128),    256, 0, stream>>>(x, xdbl, Alog, Dp, delta, ybf);
    s6_out  <<<dim3(8, 32),  256, 0, stream>>>(ybf, WoutT, bout, out);
}
